// MyLSTM_66468913873072
// MI455X (gfx1250) — matrix operand order repaired and checked by reading, not compiled, not run on hardware
//
#include <hip/hip_runtime.h>
#include <hip/hip_bf16.h>

typedef __attribute__((ext_vector_type(16))) __bf16 v16bf;
typedef __attribute__((ext_vector_type(8)))  __bf16 v8bf;
typedef __attribute__((ext_vector_type(8)))  float  v8f;
typedef __attribute__((ext_vector_type(4)))  int    v4i;

#define T_STEPS 2048
#define BATCH   256
#define IDIM    512
#define HDIM    512
#define KDIM    1024      /* I + H */
#define G4H     2048      /* 4*H   */
#define NBLOCKS 64
#define NTHREADS 256
#define SROW    520       /* 512 + 8 bf16 pad: rows staggered 4 LDS banks, 16B aligned */

__device__ __forceinline__ float fsig(float x) { return 1.0f / (1.0f + __expf(-x)); }
__device__ __forceinline__ float ftanh_(float x) {
#if __has_builtin(__builtin_amdgcn_tanhf)
  return __builtin_amdgcn_tanhf(x);
#else
  return tanhf(x);
#endif
}

// ---- CDNA5 async global->LDS copy (ASYNCcnt-tracked), with sync fallback ----
#if __has_builtin(__builtin_amdgcn_global_load_async_to_lds_b128)
#define ASYNC_LDS 1
#else
#define ASYNC_LDS 0
#endif

typedef __attribute__((address_space(1))) v4i GV4;   // global int4
typedef __attribute__((address_space(3))) v4i LV4;   // LDS int4

__device__ __forceinline__ void cp16(__bf16* dst_lds, const __bf16* src_g) {
#if ASYNC_LDS
  __builtin_amdgcn_global_load_async_to_lds_b128(
      (GV4*)(__bf16*)src_g, (LV4*)dst_lds, 0, 0);
#else
  *(v8bf*)dst_lds = *(const v8bf*)src_g;
#endif
}

__device__ __forceinline__ void wait_async_le4() {
#if ASYNC_LDS
#if __has_builtin(__builtin_amdgcn_s_wait_asynccnt)
  __builtin_amdgcn_s_wait_asynccnt(4);
#else
  asm volatile("s_wait_asynccnt 0x4" ::: "memory");
#endif
#endif
}

// ---------------- init: fused bias, zero c / h0 / barrier counter ----------------
__global__ void k_init(const float* __restrict__ b_ih, const float* __restrict__ b_hh,
                       float* __restrict__ bias, float* __restrict__ cstate,
                       __bf16* __restrict__ h0, unsigned int* __restrict__ bar) {
  int i = blockIdx.x * blockDim.x + threadIdx.x;
  if (i < G4H) bias[i] = b_ih[i] + b_hh[i];
  if (i < BATCH * HDIM) { cstate[i] = 0.0f; h0[i] = (__bf16)0.0f; }
  if (i == 0) *bar = 0u;
}

// ---------------- x: fp32 -> bf16 ----------------
__global__ void k_cvt_x(const float* __restrict__ x, __bf16* __restrict__ xb, size_t n4) {
  size_t i = ((size_t)blockIdx.x * blockDim.x + threadIdx.x);
  if (i < n4) {
    float4 v = *(const float4*)(x + i * 4);
    size_t o = i * 4;
    xb[o + 0] = (__bf16)v.x; xb[o + 1] = (__bf16)v.y;
    xb[o + 2] = (__bf16)v.z; xb[o + 3] = (__bf16)v.w;
  }
}

// ---------------- pack [W_ih ; W_hh] into WMMA B-operand tile order ----------------
// wpack layout: [ntile(128)][ktile(32)][lane(32)][e(16)] bf16
// B 32x16 bf16 layout: lane<16 -> K = ktile*32 + (e<8 ? e : 16+e-8), N = lane;  lane>=16 -> K += 8, N = lane-16
__global__ void k_pack_w(const float* __restrict__ wih, const float* __restrict__ whh,
                         __bf16* __restrict__ wpack) {
  int idx = blockIdx.x * blockDim.x + threadIdx.x;
  if (idx >= 128 * 32 * 32 * 16) return;
  int e     = idx & 15;
  int lane  = (idx >> 4) & 31;
  int kt    = (idx >> 9) & 31;
  int ntile = idx >> 14;
  int nrow  = ntile * 16 + (lane & 15);                 // row of stacked [4H, I+H]
  int k     = kt * 32 + ((lane < 16) ? 0 : 8) + (e & 7) + ((e >> 3) << 4);     // 0..1023
  float val = (k < IDIM) ? wih[(size_t)nrow * IDIM + k]
                         : whh[(size_t)nrow * HDIM + (k - IDIM)];
  wpack[idx] = (__bf16)val;
}

// ---------------- persistent recurrent kernel ----------------
// grid = 64 WGs x 256 thr (8 waves). Wave owns (mblk, hn): 16 batch rows x 16 hidden cols,
// computes all 4 gate tiles (128 WMMA/step) and the LSTM cell update locally.
// x_{t+1} is async-prefetched into a double-buffered LDS tile during step t's compute.
__global__ __launch_bounds__(NTHREADS) void lstm_persistent(
    const __bf16* __restrict__ xbf,    // [T,B,I] bf16
    const __bf16* __restrict__ wpack,  // packed weights
    const float*  __restrict__ bias,   // [4H] = b_ih + b_hh
    float*        __restrict__ cstate, // [B,H] fp32
    const __bf16* __restrict__ h0,     // [B,H] zeros
    __bf16*       __restrict__ hs,     // [T,B,H] bf16
    unsigned int* __restrict__ bar) {
  __shared__ __align__(16) __bf16 sX[2][16 * SROW];   // x_t rows, double-buffered
  __shared__ __align__(16) __bf16 sH[16 * SROW];      // h_{t-1} rows

  const int tid  = threadIdx.x;
  const int wave = tid >> 5;
  const int lane = tid & 31;
  const int mblk = blockIdx.x >> 2;                      // 0..15  (batch tile)
  const int hn   = ((blockIdx.x & 3) << 3) + wave;       // 0..31  (hidden tile)
  const int n    = lane & 15;
  const int mAdd = (lane >> 4) << 3;                     // C/D layout: lanes 16-31 -> M+8
  const int col  = hn * 16 + n;
  const float bi  = bias[col];
  const float bff = bias[512 + col];
  const float bg  = bias[1024 + col];
  const float bo  = bias[1536 + col];

  const int srow = tid >> 4;     // staging: 16 threads per A row, 4 x 16B chunks each
  const int scb  = tid & 15;

  const int abase = (lane < 16) ? 0 : 8;                 // A-operand K sub-offset
  const int rowa  = (lane & 15) * SROW;

  // prologue: async prefetch x_0 into buffer 0
  {
    const __bf16* xr = xbf + (size_t)(mblk * 16 + srow) * IDIM;
    __bf16* xd = &sX[0][srow * SROW];
#pragma unroll
    for (int i = 0; i < 4; ++i) {
      int ch = (scb + i * 16) * 8;
      cp16(xd + ch, xr + ch);
    }
  }

  for (int t = 0; t < T_STEPS; ++t) {
    // ---- async stage h_{t-1} -> sH;  async prefetch x_{t+1} -> sX[(t+1)&1] ----
    {
      const __bf16* hr = ((t == 0) ? h0 : hs + (size_t)(t - 1) * BATCH * HDIM)
                         + (size_t)(mblk * 16 + srow) * HDIM;
      __bf16* hd = &sH[srow * SROW];
#pragma unroll
      for (int i = 0; i < 4; ++i) {
        int ch = (scb + i * 16) * 8;
        cp16(hd + ch, hr + ch);
      }
      int tn = (t + 1 < T_STEPS) ? t + 1 : t;            // clamped dummy on last step
      const __bf16* xr = xbf + ((size_t)tn * BATCH + mblk * 16 + srow) * IDIM;
      __bf16* xd = &sX[(t + 1) & 1][srow * SROW];
#pragma unroll
      for (int i = 0; i < 4; ++i) {
        int ch = (scb + i * 16) * 8;
        cp16(xd + ch, xr + ch);
      }
    }
    // async loads complete in order: <=4 outstanding drains x_t + h_{t-1},
    // leaves the x_{t+1} prefetch in flight.
    wait_async_le4();
    __syncthreads();

    // ---- gates[m, g*512 + hn*16 + n] via v_wmma_f32_16x16x32_bf16, K = 1024 ----
    // Two straight loops (x-half, h-half): A pointer is a linear induction, no selects.
    v8f acc0 = {}, acc1 = {}, acc2 = {}, acc3 = {};
    const __bf16* aX = &sX[t & 1][rowa] + abase;
    const __bf16* aH = &sH[rowa] + abase;
    const size_t lb = (size_t)lane * 16;

#pragma unroll 4
    for (int kt = 0; kt < 16; ++kt) {
      v16bf a;
      v8bf lo = *(const v8bf*)(aX + kt * 32);
      v8bf hi = *(const v8bf*)(aX + kt * 32 + 16);
#pragma unroll
      for (int e = 0; e < 8; ++e) { a[e] = lo[e]; a[8 + e] = hi[e]; }

      v16bf b0 = *(const v16bf*)(wpack + ((size_t)((0 * 32 + hn) * 32 + kt) * 512) + lb);
      v16bf b1 = *(const v16bf*)(wpack + ((size_t)((1 * 32 + hn) * 32 + kt) * 512) + lb);
      v16bf b2 = *(const v16bf*)(wpack + ((size_t)((2 * 32 + hn) * 32 + kt) * 512) + lb);
      v16bf b3 = *(const v16bf*)(wpack + ((size_t)((3 * 32 + hn) * 32 + kt) * 512) + lb);

      acc0 = __builtin_amdgcn_wmma_f32_16x16x32_bf16(false, a, false, b0, (short)0, acc0, false, false);
      acc1 = __builtin_amdgcn_wmma_f32_16x16x32_bf16(false, a, false, b1, (short)0, acc1, false, false);
      acc2 = __builtin_amdgcn_wmma_f32_16x16x32_bf16(false, a, false, b2, (short)0, acc2, false, false);
      acc3 = __builtin_amdgcn_wmma_f32_16x16x32_bf16(false, a, false, b3, (short)0, acc3, false, false);
    }

#pragma unroll 4
    for (int kt = 0; kt < 16; ++kt) {
      v16bf a;
      v8bf lo = *(const v8bf*)(aH + kt * 32);
      v8bf hi = *(const v8bf*)(aH + kt * 32 + 16);
#pragma unroll
      for (int e = 0; e < 8; ++e) { a[e] = lo[e]; a[8 + e] = hi[e]; }

      v16bf b0 = *(const v16bf*)(wpack + ((size_t)((0 * 32 + hn) * 32 + kt + 16) * 512) + lb);
      v16bf b1 = *(const v16bf*)(wpack + ((size_t)((1 * 32 + hn) * 32 + kt + 16) * 512) + lb);
      v16bf b2 = *(const v16bf*)(wpack + ((size_t)((2 * 32 + hn) * 32 + kt + 16) * 512) + lb);
      v16bf b3 = *(const v16bf*)(wpack + ((size_t)((3 * 32 + hn) * 32 + kt + 16) * 512) + lb);

      acc0 = __builtin_amdgcn_wmma_f32_16x16x32_bf16(false, a, false, b0, (short)0, acc0, false, false);
      acc1 = __builtin_amdgcn_wmma_f32_16x16x32_bf16(false, a, false, b1, (short)0, acc1, false, false);
      acc2 = __builtin_amdgcn_wmma_f32_16x16x32_bf16(false, a, false, b2, (short)0, acc2, false, false);
      acc3 = __builtin_amdgcn_wmma_f32_16x16x32_bf16(false, a, false, b3, (short)0, acc3, false, false);
    }

    // ---- LSTM cell update (wave-local: this wave owns these 16x16 c/h elements) ----
    {
      __bf16* hout = hs + (size_t)t * BATCH * HDIM;
#pragma unroll
      for (int v = 0; v < 8; ++v) {
        int b = mblk * 16 + v + mAdd;
        size_t idx = (size_t)b * HDIM + col;
        float iv = fsig(acc0[v] + bi);
        float fv = fsig(acc1[v] + bff);
        float gv = ftanh_(acc2[v] + bg);
        float ov = fsig(acc3[v] + bo);
        float cn = fv * cstate[idx] + iv * gv;
        cstate[idx] = cn;
        hout[idx] = (__bf16)(ov * ftanh_(cn));
      }
    }

    // ---- grid-wide generation barrier: h_t visible to all WGs before t+1 ----
    __syncthreads();
    if (tid == 0) {
      __hip_atomic_fetch_add(bar, 1u, __ATOMIC_RELEASE, __HIP_MEMORY_SCOPE_AGENT);
      unsigned target = (unsigned)(t + 1) * NBLOCKS;
      while (__hip_atomic_load(bar, __ATOMIC_ACQUIRE, __HIP_MEMORY_SCOPE_AGENT) < target) {
        __builtin_amdgcn_s_sleep(2);
      }
    }
    __syncthreads();
  }
}

// ---------------- final projection: out[t,b] = hs[t,b,:] . W_out + b_out ----------------
__global__ __launch_bounds__(256) void k_out(const __bf16* __restrict__ hs,
                                             const float* __restrict__ wout,
                                             const float* __restrict__ bout,
                                             float* __restrict__ out) {
  size_t row = (size_t)blockIdx.x * 8 + (threadIdx.x >> 5);   // one wave per (t,b)
  int lane = threadIdx.x & 31;
  const __bf16* hr = hs + row * HDIM + lane * 16;
  const float*  wr = wout + lane * 16;
  float s = 0.0f;
#pragma unroll
  for (int j = 0; j < 16; ++j) s += (float)hr[j] * wr[j];
#pragma unroll
  for (int off = 16; off > 0; off >>= 1) s += __shfl_xor(s, off, 32);
  if (lane == 0) out[row] = s + bout[0];
}

extern "C" void kernel_launch(void* const* d_in, const int* in_sizes, int n_in,
                              void* d_out, int out_size, void* d_ws, size_t ws_size,
                              hipStream_t stream) {
  (void)in_sizes; (void)n_in; (void)out_size; (void)ws_size;
  const float* x    = (const float*)d_in[0];
  const float* wih  = (const float*)d_in[1];
  const float* whh  = (const float*)d_in[2];
  const float* bih  = (const float*)d_in[3];
  const float* bhh  = (const float*)d_in[4];
  const float* wout = (const float*)d_in[5];
  const float* bout = (const float*)d_in[6];
  float* out = (float*)d_out;

  char* ws = (char*)d_ws;
  const size_t SZ_XB    = (size_t)T_STEPS * BATCH * IDIM * 2;   // 512 MB
  const size_t SZ_HS    = (size_t)T_STEPS * BATCH * HDIM * 2;   // 512 MB
  const size_t SZ_WP    = (size_t)G4H * KDIM * 2;               // 4 MB
  const size_t SZ_BIAS  = (size_t)G4H * 4;
  const size_t SZ_C     = (size_t)BATCH * HDIM * 4;
  const size_t SZ_H0    = (size_t)BATCH * HDIM * 2;

  size_t off = 0;
  __bf16* xb    = (__bf16*)(ws + off); off += SZ_XB;
  __bf16* hsbuf = (__bf16*)(ws + off); off += SZ_HS;
  __bf16* wpack = (__bf16*)(ws + off); off += SZ_WP;
  float*  bias  = (float*)(ws + off);  off += SZ_BIAS;
  float*  cst   = (float*)(ws + off);  off += SZ_C;
  __bf16* h0    = (__bf16*)(ws + off); off += SZ_H0;
  unsigned int* bar = (unsigned int*)(ws + off); off += 256;

  k_init<<<(BATCH * HDIM + 255) / 256, 256, 0, stream>>>(bih, bhh, bias, cst, h0, bar);

  size_t n4 = (size_t)T_STEPS * BATCH * IDIM / 4;
  k_cvt_x<<<(unsigned)((n4 + 255) / 256), 256, 0, stream>>>(x, xb, n4);

  k_pack_w<<<(128 * 32 * 32 * 16) / 256, 256, 0, stream>>>(wih, whh, wpack);

  lstm_persistent<<<NBLOCKS, NTHREADS, 0, stream>>>(xb, wpack, bias, cst, h0, hsbuf, bar);

  k_out<<<(T_STEPS * BATCH) / 8, 256, 0, stream>>>(hsbuf, wout, bout, out);
}
